// VertixRefineShapeNet_18880676233658
// MI455X (gfx1250) — matrix operand order repaired and checked by reading, not compiled, not run on hardware
//
#include <hip/hip_runtime.h>

// ---------------------------------------------------------------------------
// VertixRefineShapeNet for MI455X (gfx1250, wave32, WMMA bf16 16x16x32 + TDM)
// ---------------------------------------------------------------------------

typedef __attribute__((ext_vector_type(16))) __bf16    v16bf;
typedef __attribute__((ext_vector_type(8)))  __bf16    v8bf;
typedef __attribute__((ext_vector_type(8)))  float     v8f;
typedef __attribute__((ext_vector_type(4)))  unsigned  v4u;
typedef __attribute__((ext_vector_type(8)))  int       v8i;
typedef __attribute__((ext_vector_type(4)))  int       v4i;

#define BATCH   8
#define NV      2562
#define NPAD    2576          // rows padded to multiple of 16
#define MT      (NPAD / 16)   // 161 M-tiles
#define KADJ    2592          // adjacency K padded to multiple of 32
#define C_X0    288           // gc0 input cols (259 -> 288)
#define C_X12   160           // gc1/gc2 input cols (131 -> 160)
#define C_XW0   144           // xw0 scratch leading dim
#define KFEAT   3840
#define KC      96            // adjacency-GEMM B chunk (KADJ = 27 * 96)
#define NCHUNK  (KADJ / KC)   // 27
#define ADJ_BLK 21            // ceil(MT / 8) M-tile groups per batch

__device__ inline v8f vzero8() {
  v8f z;
#pragma unroll
  for (int i = 0; i < 8; ++i) z[i] = 0.0f;
  return z;
}

// ---------------------------------------------------------------------------
// TDM: issue a 2D bf16 tile load (rows x KC, row stride KADJ elems) into LDS.
// D# layout per CDNA5 ISA ch.8 (group0: count/lds/global/type, group1: dims).
// ---------------------------------------------------------------------------
__device__ inline void tdm_load_tile(const __bf16* gptr, unsigned lds_off,
                                     int rows, int tile_k, int row_stride) {
  unsigned long long ga = (unsigned long long)(uintptr_t)gptr;
  v4u g0;
  g0.x = 1u;                                           // count = 1 (valid D#)
  g0.y = lds_off;                                      // lds_addr (bytes)
  g0.z = (unsigned)(ga & 0xFFFFFFFFu);                 // global_addr[31:0]
  g0.w = (unsigned)((ga >> 32) & 0x01FFFFFFu)          // global_addr[56:32]
         | (2u << 30);                                 // type = 2 ("image")
  v8i g1;
  g1[0] = (1 << 16);                                   // data_size = 1 (2 bytes)
  g1[1] = (int)((row_stride & 0xFFFF) << 16);          // tensor_dim0[15:0]
  g1[2] = (int)(((unsigned)row_stride >> 16) | (((unsigned)rows & 0xFFFF) << 16)); // dim0 hi / dim1 lo
  g1[3] = (int)(((unsigned)tile_k & 0xFFFF) << 16);    // dim1 hi = 0, tile_dim0 = tile_k
  g1[4] = rows;                                        // tile_dim1 = rows, tile_dim2 = 0
  g1[5] = row_stride;                                  // tensor_dim0_stride[31:0]
  g1[6] = 0;                                           // stride hi / dim1_stride lo
  g1[7] = 0;
  v4i g2; v4i g3;
#pragma unroll
  for (int i = 0; i < 4; ++i) { g2[i] = 0; g3[i] = 0; }
#if defined(__clang_major__) && (__clang_major__ >= 23)
  v8i g4;
#pragma unroll
  for (int i = 0; i < 8; ++i) g4[i] = 0;
  __builtin_amdgcn_tensor_load_to_lds(g0, g1, g2, g3, g4, 0);
#else
  __builtin_amdgcn_tensor_load_to_lds(g0, g1, g2, g3, 0);
#endif
}

// ---------------------------------------------------------------------------
// Generic bf16 GEMM (small GEMMs: x@w0, x@w1).  B given transposed [N x K].
// One wave computes a 16 x (NT*16) tile with v_wmma_f32_16x16x32_bf16.
// mode: 0 = f32 row-major out, 1 = bf16 row-major out, 2 = bf16 transposed out
// ---------------------------------------------------------------------------
template <int NT>
__global__ __launch_bounds__(256) void gemm_bf16_kernel(
    const __bf16* __restrict__ A, size_t a_bs, int lda,
    const __bf16* __restrict__ BT, size_t bt_bs, int ldbt,
    int mtiles, int ksteps,
    const float* __restrict__ Cadd, size_t c_bs, int ldc,
    int mode, int relu, int mvalid, int ncols,
    void* __restrict__ Out, size_t o_bs, int ldo, int col_off,
    int total_tiles)
{
  const int wid = (int)((blockIdx.x * blockDim.x + threadIdx.x) >> 5);
  if (wid >= total_tiles) return;               // wave-uniform: EXEC stays all-1s
  const int lane  = threadIdx.x & 31;
  const int b     = wid / mtiles;
  const int mt    = wid % mtiles;
  const int lhalf = lane >> 4;
  const int l16   = lane & 15;

  const __bf16* Ap = A + (size_t)b * a_bs + (size_t)(mt * 16 + l16) * lda + lhalf * 16;
  const __bf16* Bp = BT + (size_t)b * bt_bs + (size_t)l16 * ldbt + lhalf * 16;

  v8f acc[NT];
#pragma unroll
  for (int j = 0; j < NT; ++j) acc[j] = vzero8();

  for (int ks = 0; ks < ksteps; ++ks) {
    const __bf16* ap = Ap + ks * 32;
    __builtin_prefetch((const void*)(ap + 128), 0, 1);   // global_prefetch_b8
    v8bf alo = *(const v8bf*)(ap);
    v8bf ahi = *(const v8bf*)(ap + 8);
    v16bf afrag;
#pragma unroll
    for (int i = 0; i < 8; ++i) { afrag[i] = alo[i]; afrag[i + 8] = ahi[i]; }

    // Batch all B-fragment loads first so they issue back-to-back,
    // then drain with the WMMA chain (hides load latency).
    v16bf bfr[NT];
#pragma unroll
    for (int j = 0; j < NT; ++j)
      bfr[j] = *(const v16bf*)(Bp + (size_t)(j * 16) * ldbt + ks * 32);
#pragma unroll
    for (int j = 0; j < NT; ++j)
      acc[j] = __builtin_amdgcn_wmma_f32_16x16x32_bf16(
          false, afrag, false, bfr[j], (short)0, acc[j], false, false);
  }

  const int mbase = mt * 16 + lhalf * 8;
#pragma unroll
  for (int j = 0; j < NT; ++j) {
    const int n = j * 16 + l16;
#pragma unroll
    for (int r = 0; r < 8; ++r) {
      const int m = mbase + r;
      if (m < mvalid && n < ncols) {
        float v = acc[j][r];
        if (Cadd) v += Cadd[(size_t)b * c_bs + (size_t)m * ldc + n];
        if (relu) v = fmaxf(v, 0.0f);
        if (mode == 0) {
          ((float*)Out)[(size_t)b * o_bs + (size_t)m * ldo + col_off + n] = v;
        } else if (mode == 1) {
          ((__bf16*)Out)[(size_t)b * o_bs + (size_t)m * ldo + col_off + n] = (__bf16)v;
        } else {
          ((__bf16*)Out)[(size_t)b * o_bs + (size_t)n * ldo + m] = (__bf16)v;
        }
      }
    }
  }
}

// ---------------------------------------------------------------------------
// Adjacency GEMM:  out = epilogue(adj[b] @ xw1[b])
// 8 waves/block, one 16x128 tile strip per wave.  B (128 x KADJ, transposed
// layout) is staged through LDS in KC-wide chunks by the Tensor Data Mover,
// double-buffered; A (adjacency, 107 MB bf16) streams straight from HBM.
// ---------------------------------------------------------------------------
__global__ __launch_bounds__(256) void adj_gemm_lds_kernel(
    const __bf16* __restrict__ A, size_t a_bs,
    const __bf16* __restrict__ BT, size_t bt_bs,
    const float* __restrict__ Cadd, size_t c_bs, int ldc,
    int mode, int relu, int col_off,
    void* __restrict__ Out, size_t o_bs, int ldo)
{
  __shared__ __align__(32) __bf16 Bs[2][128 * KC];   // 2 x 24 KB

  const int b     = blockIdx.x / ADJ_BLK;
  const int mtblk = blockIdx.x % ADJ_BLK;
  const int wave  = threadIdx.x >> 5;
  const int lane  = threadIdx.x & 31;
  const int mt    = mtblk * 8 + wave;
  const bool active = (mt < MT);                      // wave-uniform
  const int mts   = active ? mt : (MT - 1);
  const int lhalf = lane >> 4;
  const int l16   = lane & 15;

  const __bf16* Ap = A + (size_t)b * a_bs + (size_t)(mts * 16 + l16) * KADJ + lhalf * 16;
  const __bf16* Bg = BT + (size_t)b * bt_bs;

  v8f acc[8];
#pragma unroll
  for (int j = 0; j < 8; ++j) acc[j] = vzero8();

  if (wave == 0)
    tdm_load_tile(Bg, (unsigned)(uintptr_t)&Bs[0][0], 128, KC, KADJ);

  for (int c = 0; c < NCHUNK; ++c) {
    if (wave == 0) {
      if (c + 1 < NCHUNK) {
        tdm_load_tile(Bg + (size_t)(c + 1) * KC,
                      (unsigned)(uintptr_t)&Bs[(c + 1) & 1][0], 128, KC, KADJ);
        __builtin_amdgcn_s_wait_tensorcnt(1);          // chunk c complete (in-order)
      } else {
        __builtin_amdgcn_s_wait_tensorcnt(0);          // last chunk complete
      }
    }
    __syncthreads();                                   // chunk c visible to all waves

    if (active) {
      const __bf16* Bl = &Bs[c & 1][0] + (size_t)l16 * KC + lhalf * 16;
#pragma unroll
      for (int ck = 0; ck < KC / 32; ++ck) {
        const __bf16* ap = Ap + (c * KC + ck * 32);
        __builtin_prefetch((const void*)(ap + 256), 0, 1);
        v8bf alo = *(const v8bf*)(ap);
        v8bf ahi = *(const v8bf*)(ap + 8);
        v16bf afrag;
#pragma unroll
        for (int i = 0; i < 8; ++i) { afrag[i] = alo[i]; afrag[i + 8] = ahi[i]; }

        // Batch the 16 ds_load_b128 issues, then drain with the WMMA chain.
        v16bf bfr[8];
#pragma unroll
        for (int j = 0; j < 8; ++j)
          bfr[j] = *(const v16bf*)(Bl + (size_t)(j * 16) * KC + ck * 32);
#pragma unroll
        for (int j = 0; j < 8; ++j)
          acc[j] = __builtin_amdgcn_wmma_f32_16x16x32_bf16(
              false, afrag, false, bfr[j], (short)0, acc[j], false, false);
      }
    }
    __syncthreads();                                   // all reads of buf done
  }

  if (active) {
    const int mbase = mt * 16 + lhalf * 8;
#pragma unroll
    for (int j = 0; j < 8; ++j) {
      const int n = j * 16 + l16;
#pragma unroll
      for (int r = 0; r < 8; ++r) {
        const int m = mbase + r;
        if (m < NV) {
          float v = acc[j][r];
          v += Cadd[(size_t)b * c_bs + (size_t)m * ldc + n];
          if (relu) v = fmaxf(v, 0.0f);
          if (mode == 0)
            ((float*)Out)[(size_t)b * o_bs + (size_t)m * ldo + col_off + n] = v;
          else
            ((__bf16*)Out)[(size_t)b * o_bs + (size_t)m * ldo + col_off + n] = (__bf16)v;
        }
      }
    }
  }
}

// ---------------------------------------------------------------------------
// Fused vert_align + projection GEMM (K=3840 gathered bilinear taps -> 128)
// ---------------------------------------------------------------------------
struct Smp { int i00, i01, i10, i11, hw; float w00, w01, w10, w11; };

__device__ inline Smp make_smp(int b, float px, float py, int C, int H, int W) {
  float x = (fminf(fmaxf(px, -1.0f), 1.0f) + 1.0f) * 0.5f * (float)(W - 1);
  float y = (fminf(fmaxf(py, -1.0f), 1.0f) + 1.0f) * 0.5f * (float)(H - 1);
  float x0f = floorf(x), y0f = floorf(y);
  int x0 = (int)x0f, y0 = (int)y0f;
  int x1 = min(x0 + 1, W - 1), y1 = min(y0 + 1, H - 1);
  float wx = x - x0f, wy = y - y0f;
  Smp s;
  int base = b * C * H * W;
  s.i00 = base + y0 * W + x0;
  s.i01 = base + y0 * W + x1;
  s.i10 = base + y1 * W + x0;
  s.i11 = base + y1 * W + x1;
  s.hw  = H * W;
  s.w00 = (1.0f - wx) * (1.0f - wy);
  s.w01 = wx * (1.0f - wy);
  s.w10 = (1.0f - wx) * wy;
  s.w11 = wx * wy;
  return s;
}

__device__ inline float fetch4(const float* __restrict__ f, const Smp& s, int c) {
  int o = c * s.hw;
  return f[s.i00 + o] * s.w00 + f[s.i01 + o] * s.w01 +
         f[s.i10 + o] * s.w10 + f[s.i11 + o] * s.w11;
}

__global__ __launch_bounds__(256) void vert_project_kernel(
    const float* __restrict__ c2, const float* __restrict__ c3,
    const float* __restrict__ c4, const float* __restrict__ c5,
    const float* __restrict__ pos, const float* __restrict__ bias,
    const __bf16* __restrict__ WT, __bf16* __restrict__ X0, int total_tiles)
{
  const int wid = (int)((blockIdx.x * blockDim.x + threadIdx.x) >> 5);
  if (wid >= total_tiles) return;
  const int lane  = threadIdx.x & 31;
  const int b     = wid / MT;
  const int mt    = wid % MT;
  const int lhalf = lane >> 4;
  const int l16   = lane & 15;
  const int m     = mt * 16 + l16;
  const int v     = min(m, NV - 1);

  const float px = pos[((size_t)b * NV + v) * 3 + 0];
  const float py = pos[((size_t)b * NV + v) * 3 + 1];
  Smp s0 = make_smp(b, px, py, 256,  56, 56);
  Smp s1 = make_smp(b, px, py, 512,  28, 28);
  Smp s2 = make_smp(b, px, py, 1024, 14, 14);
  Smp s3 = make_smp(b, px, py, 2048,  7,  7);

  const __bf16* Bp = WT + (size_t)l16 * KFEAT + lhalf * 16;
  const int kb = lhalf * 16;

  v8f acc[8];
#pragma unroll
  for (int j = 0; j < 8; ++j) acc[j] = vzero8();

  for (int ks = 0; ks < KFEAT / 32; ++ks) {
    v16bf afrag;
#pragma unroll
    for (int i = 0; i < 16; ++i) {
      const int k = ks * 32 + kb + i;
      float val;
      if (k < 256)       val = fetch4(c2, s0, k);
      else if (k < 768)  val = fetch4(c3, s1, k - 256);
      else if (k < 1792) val = fetch4(c4, s2, k - 768);
      else               val = fetch4(c5, s3, k - 1792);
      afrag[i] = (__bf16)val;
    }
    v16bf bfr[8];
#pragma unroll
    for (int j = 0; j < 8; ++j)
      bfr[j] = *(const v16bf*)(Bp + (size_t)(j * 16) * KFEAT + ks * 32);
#pragma unroll
    for (int j = 0; j < 8; ++j)
      acc[j] = __builtin_amdgcn_wmma_f32_16x16x32_bf16(
          false, afrag, false, bfr[j], (short)0, acc[j], false, false);
  }

  const int mbase = mt * 16 + lhalf * 8;
#pragma unroll
  for (int j = 0; j < 8; ++j) {
    const int n = j * 16 + l16;
#pragma unroll
    for (int r = 0; r < 8; ++r) {
      const int mm = mbase + r;
      if (mm < NV) {
        float vv = acc[j][r] + bias[n];
        X0[((size_t)b * NPAD + mm) * C_X0 + 131 + n] = (__bf16)vv;
      }
    }
  }
}

// ---------------------------------------------------------------------------
// Helper / elementwise kernels
// ---------------------------------------------------------------------------
__global__ void convT_kernel(const float* __restrict__ W, __bf16* __restrict__ WT,
                             int K, int N, int Kpad, int Npad)
{
  int t = blockIdx.x * blockDim.x + threadIdx.x;
  if (t >= Npad * Kpad) return;
  int n = t / Kpad, k = t % Kpad;
  WT[t] = (k < K && n < N) ? (__bf16)W[(size_t)k * N + n] : (__bf16)0.0f;
}

__global__ void adj_bf16_kernel(const float* __restrict__ adj,
                                __bf16* __restrict__ out, long long total8)
{
  long long t = (long long)blockIdx.x * blockDim.x + threadIdx.x;
  if (t >= total8) return;
  long long e = t * 8;                                // KADJ % 8 == 0: no row straddle
  const long long per = (long long)NPAD * KADJ;
  int b = (int)(e / per);
  long long rem = e % per;
  int r = (int)(rem / KADJ), c0 = (int)(rem % KADJ);
  v8bf v;
  if (r < NV && (c0 + 7) < NV) {
    const float* src = adj + ((size_t)b * NV + r) * NV + c0;
#pragma unroll
    for (int i = 0; i < 8; ++i) v[i] = (__bf16)src[i];
  } else {
#pragma unroll
    for (int i = 0; i < 8; ++i) {
      int c = c0 + i;
      float f = (r < NV && c < NV) ? adj[((size_t)b * NV + r) * NV + c] : 0.0f;
      v[i] = (__bf16)f;
    }
  }
  *(v8bf*)(out + e) = v;
}

__global__ void pack_kernel(const float* __restrict__ feat, const float* __restrict__ pos,
                            __bf16* __restrict__ X0, __bf16* __restrict__ X1,
                            __bf16* __restrict__ X2, int total)
{
  int t = blockIdx.x * blockDim.x + threadIdx.x;
  if (t >= total) return;
  int b = t / NV, n = t % NV;
  const float* f = feat + (size_t)t * 128;
  __bf16* x0 = X0 + ((size_t)b * NPAD + n) * C_X0;
#pragma unroll
  for (int c8 = 0; c8 < 16; ++c8) {
    v8bf v;
#pragma unroll
    for (int i = 0; i < 8; ++i) v[i] = (__bf16)f[c8 * 8 + i];
    *(v8bf*)(x0 + c8 * 8) = v;                         // X0 row is 16B-aligned
  }
  float p0 = pos[(size_t)t * 3 + 0];
  float p1 = pos[(size_t)t * 3 + 1];
  float p2 = pos[(size_t)t * 3 + 2];
  x0[128] = (__bf16)p0; x0[129] = (__bf16)p1; x0[130] = (__bf16)p2;
  __bf16* x1 = X1 + ((size_t)b * NPAD + n) * C_X12;
  x1[0] = (__bf16)p0; x1[1] = (__bf16)p1; x1[2] = (__bf16)p2;
  __bf16* x2 = X2 + ((size_t)b * NPAD + n) * C_X12;
  x2[0] = (__bf16)p0; x2[1] = (__bf16)p1; x2[2] = (__bf16)p2;
}

__global__ void final_pos_kernel(const float* __restrict__ pos, const float* __restrict__ nf,
                                 const float* __restrict__ w1, const float* __restrict__ b1,
                                 float* __restrict__ outpos, int total)
{
  int t = blockIdx.x * blockDim.x + threadIdx.x;
  if (t >= total) return;
  const float* f = nf + (size_t)t * 128;
  float a0 = b1[0], a1 = b1[1], a2 = b1[2];
#pragma unroll 4
  for (int c = 0; c < 128; ++c) {
    float fv = f[c];
    a0 += fv * w1[c * 3 + 0];
    a1 += fv * w1[c * 3 + 1];
    a2 += fv * w1[c * 3 + 2];
  }
  outpos[(size_t)t * 3 + 0] = pos[(size_t)t * 3 + 0] + tanhf(a0);
  outpos[(size_t)t * 3 + 1] = pos[(size_t)t * 3 + 1] + tanhf(a1);
  outpos[(size_t)t * 3 + 2] = pos[(size_t)t * 3 + 2] + tanhf(a2);
}

// ---------------------------------------------------------------------------
// Host side
// ---------------------------------------------------------------------------
static inline void launch_gemm(int NT,
    const __bf16* A, size_t a_bs, int lda,
    const __bf16* BT, size_t bt_bs, int ldbt, int ksteps,
    const float* Cadd, size_t c_bs, int ldc,
    int mode, int relu, int mvalid, int ncols,
    void* Out, size_t o_bs, int ldo, int col_off, hipStream_t stream)
{
  const int total_tiles = BATCH * MT;
  const int blocks = (total_tiles * 32 + 255) / 256;
  if (NT == 9) {
    gemm_bf16_kernel<9><<<blocks, 256, 0, stream>>>(A, a_bs, lda, BT, bt_bs, ldbt,
        MT, ksteps, Cadd, c_bs, ldc, mode, relu, mvalid, ncols,
        Out, o_bs, ldo, col_off, total_tiles);
  } else {
    gemm_bf16_kernel<8><<<blocks, 256, 0, stream>>>(A, a_bs, lda, BT, bt_bs, ldbt,
        MT, ksteps, Cadd, c_bs, ldc, mode, relu, mvalid, ncols,
        Out, o_bs, ldo, col_off, total_tiles);
  }
}

extern "C" void kernel_launch(void* const* d_in, const int* in_sizes, int n_in,
                              void* d_out, int out_size, void* d_ws, size_t ws_size,
                              hipStream_t stream) {
  const float* c2     = (const float*)d_in[0];
  const float* c3     = (const float*)d_in[1];
  const float* c4     = (const float*)d_in[2];
  const float* c5     = (const float*)d_in[3];
  const float* adj    = (const float*)d_in[4];
  const float* pos    = (const float*)d_in[5];
  const float* vfeat  = (const float*)d_in[6];
  const float* w_lin0 = (const float*)d_in[7];
  const float* b_lin0 = (const float*)d_in[8];
  const float* gc0_w0 = (const float*)d_in[9];
  const float* gc0_w1 = (const float*)d_in[10];
  const float* gc1_w0 = (const float*)d_in[11];
  const float* gc1_w1 = (const float*)d_in[12];
  const float* gc2_w0 = (const float*)d_in[13];
  const float* gc2_w1 = (const float*)d_in[14];
  const float* w_lin1 = (const float*)d_in[15];
  const float* b_lin1 = (const float*)d_in[16];

  char* ws = (char*)d_ws;
  size_t off = 0;
  auto alloc = [&](size_t bytes) -> char* {
    char* p = ws + off;
    off = (off + bytes + 255) & ~(size_t)255;
    return p;
  };

  // Zero-initialized region (padded activations) first, single memset.
  __bf16* X0   = (__bf16*)alloc((size_t)BATCH * NPAD * C_X0 * 2);
  __bf16* X1   = (__bf16*)alloc((size_t)BATCH * NPAD * C_X12 * 2);
  __bf16* X2   = (__bf16*)alloc((size_t)BATCH * NPAD * C_X12 * 2);
  __bf16* XW1T = (__bf16*)alloc((size_t)BATCH * 144 * KADJ * 2);
  const size_t zero_bytes = off;

  __bf16* ADJB  = (__bf16*)alloc((size_t)BATCH * NPAD * KADJ * 2);
  float*  XW0   = (float*) alloc((size_t)BATCH * NPAD * C_XW0 * 4);
  __bf16* WL0T  = (__bf16*)alloc((size_t)128 * KFEAT * 2);
  __bf16* G0W0T = (__bf16*)alloc((size_t)144 * 288 * 2);
  __bf16* G0W1T = (__bf16*)alloc((size_t)144 * 288 * 2);
  __bf16* G1W0T = (__bf16*)alloc((size_t)128 * 160 * 2);
  __bf16* G1W1T = (__bf16*)alloc((size_t)128 * 160 * 2);
  __bf16* G2W0T = (__bf16*)alloc((size_t)128 * 160 * 2);
  __bf16* G2W1T = (__bf16*)alloc((size_t)128 * 160 * 2);
  if (off > ws_size) return;

  (void)hipMemsetAsync(d_ws, 0, zero_bytes, stream);

  // Weight conversion (f32 -> bf16, transposed, zero-padded)
  convT_kernel<<<(128 * KFEAT + 255) / 256, 256, 0, stream>>>(w_lin0, WL0T, KFEAT, 128, KFEAT, 128);
  convT_kernel<<<(144 * 288 + 255) / 256, 256, 0, stream>>>(gc0_w0, G0W0T, 259, 131, 288, 144);
  convT_kernel<<<(144 * 288 + 255) / 256, 256, 0, stream>>>(gc0_w1, G0W1T, 259, 131, 288, 144);
  convT_kernel<<<(128 * 160 + 255) / 256, 256, 0, stream>>>(gc1_w0, G1W0T, 131, 128, 160, 128);
  convT_kernel<<<(128 * 160 + 255) / 256, 256, 0, stream>>>(gc1_w1, G1W1T, 131, 128, 160, 128);
  convT_kernel<<<(128 * 160 + 255) / 256, 256, 0, stream>>>(gc2_w0, G2W0T, 131, 128, 160, 128);
  convT_kernel<<<(128 * 160 + 255) / 256, 256, 0, stream>>>(gc2_w1, G2W1T, 131, 128, 160, 128);

  // Adjacency conversion (f32 -> padded bf16), 8 elems/thread, b128 stores
  {
    long long total8 = (long long)BATCH * NPAD * KADJ / 8;
    int blocks = (int)((total8 + 255) / 256);
    adj_bf16_kernel<<<blocks, 256, 0, stream>>>(adj, ADJB, total8);
  }

  pack_kernel<<<(BATCH * NV + 255) / 256, 256, 0, stream>>>(vfeat, pos, X0, X1, X2, BATCH * NV);

  {
    const int total_tiles = BATCH * MT;
    const int blocks = (total_tiles * 32 + 255) / 256;
    vert_project_kernel<<<blocks, 256, 0, stream>>>(c2, c3, c4, c5, pos, b_lin0, WL0T, X0, total_tiles);
  }

  const size_t aX0  = (size_t)NPAD * C_X0;
  const size_t aX12 = (size_t)NPAD * C_X12;
  const size_t aADJ = (size_t)NPAD * KADJ;
  const size_t aW0  = (size_t)NPAD * C_XW0;
  const size_t aW1T = (size_t)144 * KADJ;
  const int adj_blocks = BATCH * ADJ_BLK;

  // ---- graph conv 0 ----
  launch_gemm(9, X0, aX0, C_X0, G0W0T, 0, 288, C_X0 / 32,
              nullptr, 0, 0, 0, 0, NPAD, 144, XW0, aW0, C_XW0, 0, stream);
  launch_gemm(9, X0, aX0, C_X0, G0W1T, 0, 288, C_X0 / 32,
              nullptr, 0, 0, 2, 0, NPAD, 144, XW1T, aW1T, KADJ, 0, stream);
  adj_gemm_lds_kernel<<<adj_blocks, 256, 0, stream>>>(ADJB, aADJ, XW1T, aW1T,
              XW0, aW0, C_XW0, /*mode bf16*/1, 1, 3, X1, aX12, C_X12);

  // ---- graph conv 1 ----
  launch_gemm(8, X1, aX12, C_X12, G1W0T, 0, 160, C_X12 / 32,
              nullptr, 0, 0, 0, 0, NPAD, 128, XW0, aW0, C_XW0, 0, stream);
  launch_gemm(8, X1, aX12, C_X12, G1W1T, 0, 160, C_X12 / 32,
              nullptr, 0, 0, 2, 0, NPAD, 128, XW1T, aW1T, KADJ, 0, stream);
  adj_gemm_lds_kernel<<<adj_blocks, 256, 0, stream>>>(ADJB, aADJ, XW1T, aW1T,
              XW0, aW0, C_XW0, 1, 1, 3, X2, aX12, C_X12);

  // ---- graph conv 2 -> new_features ----
  float* out_pos  = (float*)d_out;
  float* out_feat = out_pos + (size_t)BATCH * NV * 3;
  launch_gemm(8, X2, aX12, C_X12, G2W0T, 0, 160, C_X12 / 32,
              nullptr, 0, 0, 0, 0, NPAD, 128, XW0, aW0, C_XW0, 0, stream);
  launch_gemm(8, X2, aX12, C_X12, G2W1T, 0, 160, C_X12 / 32,
              nullptr, 0, 0, 2, 0, NPAD, 128, XW1T, aW1T, KADJ, 0, stream);
  adj_gemm_lds_kernel<<<adj_blocks, 256, 0, stream>>>(ADJB, aADJ, XW1T, aW1T,
              XW0, aW0, C_XW0, /*mode f32*/0, 1, 0, out_feat, (size_t)NV * 128, 128);

  final_pos_kernel<<<(BATCH * NV + 255) / 256, 256, 0, stream>>>(
      pos, out_feat, w_lin1, b_lin1, out_pos, BATCH * NV);
}
